// Non_AR_23261542875381
// MI455X (gfx1250) — matrix operand order repaired and checked by reading, not compiled, not run on hardware
//
#include <hip/hip_runtime.h>

#define B_    32
#define N_    184
#define HIST_ 24
#define PRED_ 24
#define HID_  32
#define DFF_  128
#define LMAX  48

typedef __attribute__((ext_vector_type(16))) _Float16 v16h;
typedef __attribute__((ext_vector_type(8)))  float    v8f;

// Fast hardware transcendental approximations (v_rcp_f32 / v_rsq_f32, TRANS unit)
__device__ __forceinline__ float fast_rcp(float x)  { return __builtin_amdgcn_rcpf(x); }
__device__ __forceinline__ float fast_rsqrt(float x){ return __builtin_amdgcn_rsqf(x); }

// ---- WMMA fragment helpers (gfx1250 wave32 layouts, CDNA5 ISA §7.12.2) ----

// A-matrix 16x32 f16 fragment from an LDS f16 buffer (row-major, leading dim ld).
// lane 0-15: M=lane, K-base 0; lane 16-31: M=lane-16, K-base +8.
// VGPR v in 0..3: K = 2*(v&3); VGPR v in 4..7: K += 16.
// Per lane this is two contiguous 16-byte runs -> compiler emits 2x ds_load_b128.
__device__ __forceinline__ v16h load_A_lds(const _Float16* src, int m0, int ld,
                                           int lane, int koff) {
  const _Float16* p = src + (m0 + (lane & 15)) * ld + koff + ((lane >> 4) & 1) * 8;
  v16h a;
#pragma unroll
  for (int v = 0; v < 8; ++v) {
    int k = 2 * (v & 3) + ((v >> 2) << 4);
    a[2 * v]     = p[k];
    a[2 * v + 1] = p[k + 1];
  }
  return a;
}

// B-matrix 32x16 f16 fragment from global f32 weights W (rows = out, leading dim ld = in).
// lane 0-15: N=lane, K-base 0; lane 16-31: N=lane-16, K-base +8. VGPR v: K=2(v&3)+16(v>>2).
__device__ __forceinline__ v16h load_B_w(const float* W, int n0, int ld,
                                         int lane, int koff) {
  const float* p = W + (n0 + (lane & 15)) * ld + koff + ((lane >> 4) & 1) * 8;
  v16h b;
#pragma unroll
  for (int v = 0; v < 8; ++v) {
    b[2 * v]     = (_Float16)p[2 * v + ((v >> 2) << 3)];
    b[2 * v + 1] = (_Float16)p[2 * v + 1 + ((v >> 2) << 3)];
  }
  return b;
}

__device__ __forceinline__ v8f wmma32(v16h a, v16h b, v8f c) {
  // D = A(16x32 f16) * B(32x16 f16) + C(f32)
  return __builtin_amdgcn_wmma_f32_16x16x32_f16(false, a, false, b, (short)0, c,
                                                false, false);
}

__device__ __forceinline__ float wave_sum(float v) {
#pragma unroll
  for (int off = 16; off > 0; off >>= 1) v += __shfl_xor(v, off, 32);
  return v;
}

// LayerNorm over HID=32, one lane per channel, in place on f32 LDS buffer.
// One-pass E[x^2]-mu^2 form: the two butterflies are independent -> dual-issue.
__device__ __forceinline__ void layer_norm(float* x, int L, const float* s,
                                           const float* bb, int lane) {
  float sc = s[lane], bi = bb[lane];
  for (int r = 0; r < L; ++r) {
    float v = x[r * HID_ + lane];
    float s1 = v, s2 = v * v;
#pragma unroll
    for (int off = 16; off > 0; off >>= 1) {
      s1 += __shfl_xor(s1, off, 32);
      s2 += __shfl_xor(s2, off, 32);
    }
    float mu  = s1 * (1.0f / 32.0f);
    float var = s2 * (1.0f / 32.0f) - mu * mu;
    x[r * HID_ + lane] = (v - mu) * fast_rsqrt(var + 1e-5f) * sc + bi;
  }
}

// Sinusoidal positional encoding value for (position t, channel j).
__device__ __forceinline__ float pe_val(int t, int j) {
  int i2 = j & ~1;
  float div = __expf(-(float)i2 * (9.210340371976184f / 32.0f)); // 10000^(-i2/32)
  float ang = (float)t * div;
  return (j & 1) ? __cosf(ang) : __sinf(ang);
}

__global__ __launch_bounds__(32) void pm25_transformer_ar(
    const float* __restrict__ pm25,     // (B,24,N,1)
    const float* __restrict__ feat,     // (B,48,N,7)
    const float* __restrict__ fc_in_w,  // (32,8)
    const float* __restrict__ fc_in_b,  // (32)
    const float* __restrict__ qkv_w,    // (2,96,32)
    const float* __restrict__ qkv_b,    // (2,96)
    const float* __restrict__ ao_w,     // (2,32,32)
    const float* __restrict__ ao_b,     // (2,32)
    const float* __restrict__ ff1_w,    // (2,128,32)
    const float* __restrict__ ff1_b,    // (2,128)
    const float* __restrict__ ff2_w,    // (2,32,128)
    const float* __restrict__ ff2_b,    // (2,32)
    const float* __restrict__ ln1_s, const float* __restrict__ ln1_b,
    const float* __restrict__ ln2_s, const float* __restrict__ ln2_b,
    const float* __restrict__ fc_out_w, // (1,32)
    const float* __restrict__ fc_out_b, // (1)
    float* __restrict__ out)            // (B,24,N,1)
{
  // Per-workgroup (= one wave) LDS: 46,080 bytes out of 320 KB/WGP pool.
  __shared__ float    s_mem[LMAX * HID_];  // raw embeddings + pos-enc (persist over steps)
  __shared__ float    s_x  [LMAX * HID_];  // f32 residual stream
  __shared__ _Float16 s_xh [LMAX * HID_];  // f16 A-operand staging
  __shared__ float    s_qkv[LMAX * 96];    // Q|K|V per token
  __shared__ _Float16 s_h  [LMAX * DFF_];  // FF hidden (f16)

  const int lane = threadIdx.x;            // wave32: blockDim.x == 32
  const int seq  = blockIdx.x;             // one sequence per wave
  if (seq >= B_ * N_) return;
  const int b = seq / N_;
  const int n = seq % N_;

  // Per-lane constants
  float wi[8];
#pragma unroll
  for (int c = 0; c < 8; ++c) wi[c] = fc_in_w[lane * 8 + c];
  const float bi  = fc_in_b[lane];
  const float fow = fc_out_w[lane];
  const float fob = fc_out_b[0];

  // ---- Initial memory: fc_in(concat(pm25, feat)) + pe, rows 0..23 ----
  for (int t = 0; t < HIST_; ++t) {
    float x0 = pm25[(b * HIST_ + t) * N_ + n];
    float e = bi + x0 * wi[0];
#pragma unroll
    for (int c = 0; c < 7; ++c)
      e += feat[((b * (HIST_ + PRED_) + t) * N_ + n) * 7 + c] * wi[1 + c];
    s_mem[t * HID_ + lane] = e + pe_val(t, lane);
  }
  float xn = pm25[(b * HIST_ + (HIST_ - 1)) * N_ + n]; // uniform across lanes

  const float att_scale = 0.3535533905932738f; // 1/sqrt(DH=8)

  // ==================== autoregressive loop ====================
  for (int step = 0; step < PRED_; ++step) {
    const int L = HIST_ + 1 + step; // 25..48
    const int t = HIST_ + step;

    // Append new token embedding to mem
    {
      float e = bi + xn * wi[0];
#pragma unroll
      for (int c = 0; c < 7; ++c)
        e += feat[((b * (HIST_ + PRED_) + t) * N_ + n) * 7 + c] * wi[1 + c];
      s_mem[(L - 1) * HID_ + lane] = e + pe_val(t, lane);
    }

    // x = mem (working copy)
    for (int r = 0; r < L; ++r) s_x[r * HID_ + lane] = s_mem[r * HID_ + lane];

    // -------- 2 encoder layers --------
    for (int l = 0; l < 2; ++l) {
      const float* Wqkv = qkv_w + l * 96 * HID_;
      const float* Bqkv = qkv_b + l * 96;
      const float* Wao  = ao_w  + l * HID_ * HID_;
      const float* Bao  = ao_b  + l * HID_;
      const float* Wf1  = ff1_w + l * DFF_ * HID_;
      const float* Bf1  = ff1_b + l * DFF_;
      const float* Wf2  = ff2_w + l * HID_ * DFF_;
      const float* Bf2  = ff2_b + l * HID_;

      // x -> f16
      for (int r = 0; r < L; ++r)
        s_xh[r * HID_ + lane] = (_Float16)s_x[r * HID_ + lane];

      // ---- QKV: (L x 32) @ (32 x 96), K=32 -> one WMMA per tile ----
      for (int nt = 0; nt < 6; ++nt) {
        const int n0 = nt * 16;
        v16h bf = load_B_w(Wqkv, n0, HID_, lane, 0);
        const int col = n0 + (lane & 15);
        const float bias = Bqkv[col];
        for (int m0 = 0; m0 < L; m0 += 16) {
          v16h af = load_A_lds(s_xh, m0, HID_, lane, 0);
          v8f c = {};
          c = wmma32(af, bf, c);
#pragma unroll
          for (int v = 0; v < 8; ++v) {
            int row = m0 + v + ((lane >> 4) & 1) * 8;
            s_qkv[row * 96 + col] = c[v] + bias;
          }
        }
      }

      // ---- Attention: online-softmax, 2 queries/lane, 4 heads, DH=8 ----
      for (int qi = 0; qi < 2; ++qi) {
        const int q = lane + 32 * qi;
        if (q < L) {
#pragma unroll
          for (int h = 0; h < 4; ++h) {
            const float4* qp = (const float4*)&s_qkv[q * 96 + h * 8];
            float4 q0 = qp[0], q1 = qp[1];
            float m = -1e30f, ssum = 0.0f;
            float4 a0 = {0, 0, 0, 0}, a1 = {0, 0, 0, 0};
            for (int k = 0; k < L; ++k) {
              const float4* kp = (const float4*)&s_qkv[k * 96 + 32 + h * 8];
              float4 k0 = kp[0], k1 = kp[1];
              float s = q0.x * k0.x + q0.y * k0.y + q0.z * k0.z + q0.w * k0.w +
                        q1.x * k1.x + q1.y * k1.y + q1.z * k1.z + q1.w * k1.w;
              s *= att_scale;
              float nm   = fmaxf(m, s);
              float corr = __expf(m - nm);
              float p    = __expf(s - nm);
              ssum = ssum * corr + p;
              const float4* vp = (const float4*)&s_qkv[k * 96 + 64 + h * 8];
              float4 v0 = vp[0], v1 = vp[1];
              a0.x = a0.x * corr + p * v0.x;  a0.y = a0.y * corr + p * v0.y;
              a0.z = a0.z * corr + p * v0.z;  a0.w = a0.w * corr + p * v0.w;
              a1.x = a1.x * corr + p * v1.x;  a1.y = a1.y * corr + p * v1.y;
              a1.z = a1.z * corr + p * v1.z;  a1.w = a1.w * corr + p * v1.w;
              m = nm;
            }
            float inv = fast_rcp(ssum);      // v_rcp_f32, not IEEE div chain
            _Float16* o = &s_xh[q * HID_ + h * 8];
            o[0] = (_Float16)(a0.x * inv); o[1] = (_Float16)(a0.y * inv);
            o[2] = (_Float16)(a0.z * inv); o[3] = (_Float16)(a0.w * inv);
            o[4] = (_Float16)(a1.x * inv); o[5] = (_Float16)(a1.y * inv);
            o[6] = (_Float16)(a1.z * inv); o[7] = (_Float16)(a1.w * inv);
          }
        }
      }

      // ---- Output projection + residual, then LN1 ----
      for (int nt = 0; nt < 2; ++nt) {
        const int n0 = nt * 16;
        v16h bf = load_B_w(Wao, n0, HID_, lane, 0);
        const int col = n0 + (lane & 15);
        const float bias = Bao[col];
        for (int m0 = 0; m0 < L; m0 += 16) {
          v16h af = load_A_lds(s_xh, m0, HID_, lane, 0);
          v8f c = {};
          c = wmma32(af, bf, c);
#pragma unroll
          for (int v = 0; v < 8; ++v) {
            int row = m0 + v + ((lane >> 4) & 1) * 8;
            s_x[row * HID_ + col] += c[v] + bias;
          }
        }
      }
      layer_norm(s_x, L, ln1_s + l * HID_, ln1_b + l * HID_, lane);

      // x -> f16 again for FF
      for (int r = 0; r < L; ++r)
        s_xh[r * HID_ + lane] = (_Float16)s_x[r * HID_ + lane];

      // ---- FF1: (L x 32) @ (32 x 128), ReLU -> s_h (f16) ----
      for (int nt = 0; nt < 8; ++nt) {
        const int n0 = nt * 16;
        v16h bf = load_B_w(Wf1, n0, HID_, lane, 0);
        const int col = n0 + (lane & 15);
        const float bias = Bf1[col];
        for (int m0 = 0; m0 < L; m0 += 16) {
          v16h af = load_A_lds(s_xh, m0, HID_, lane, 0);
          v8f c = {};
          c = wmma32(af, bf, c);
#pragma unroll
          for (int v = 0; v < 8; ++v) {
            int row = m0 + v + ((lane >> 4) & 1) * 8;
            s_h[row * DFF_ + col] = (_Float16)fmaxf(c[v] + bias, 0.0f);
          }
        }
      }

      // ---- FF2: (L x 128) @ (128 x 32), K=128 = 4 chained WMMAs ----
      // B fragments are m-invariant: build once per n-tile, reuse across m-tiles.
      for (int nt = 0; nt < 2; ++nt) {
        const int n0 = nt * 16;
        v16h bf0 = load_B_w(Wf2, n0, DFF_, lane, 0);
        v16h bf1 = load_B_w(Wf2, n0, DFF_, lane, 32);
        v16h bf2 = load_B_w(Wf2, n0, DFF_, lane, 64);
        v16h bf3 = load_B_w(Wf2, n0, DFF_, lane, 96);
        const int col = n0 + (lane & 15);
        const float bias = Bf2[col];
        for (int m0 = 0; m0 < L; m0 += 16) {
          v8f c = {};
          c = wmma32(load_A_lds(s_h, m0, DFF_, lane, 0),  bf0, c);
          c = wmma32(load_A_lds(s_h, m0, DFF_, lane, 32), bf1, c);
          c = wmma32(load_A_lds(s_h, m0, DFF_, lane, 64), bf2, c);
          c = wmma32(load_A_lds(s_h, m0, DFF_, lane, 96), bf3, c);
#pragma unroll
          for (int v = 0; v < 8; ++v) {
            int row = m0 + v + ((lane >> 4) & 1) * 8;
            s_x[row * HID_ + col] += c[v] + bias;
          }
        }
      }
      layer_norm(s_x, L, ln2_s + l * HID_, ln2_b + l * HID_, lane);
    } // layers

    // ---- fc_out on last token; broadcast xn; emit prediction ----
    float hv = s_x[(L - 1) * HID_ + lane] * fow;
    hv = wave_sum(hv);
    xn = hv + fob;
    if (lane == 0) out[(b * PRED_ + step) * N_ + n] = xn;
  } // steps
}

extern "C" void kernel_launch(void* const* d_in, const int* in_sizes, int n_in,
                              void* d_out, int out_size, void* d_ws, size_t ws_size,
                              hipStream_t stream) {
  (void)in_sizes; (void)n_in; (void)out_size; (void)d_ws; (void)ws_size;
  pm25_transformer_ar<<<B_ * N_, 32, 0, stream>>>(
      (const float*)d_in[0],  (const float*)d_in[1],  (const float*)d_in[2],
      (const float*)d_in[3],  (const float*)d_in[4],  (const float*)d_in[5],
      (const float*)d_in[6],  (const float*)d_in[7],  (const float*)d_in[8],
      (const float*)d_in[9],  (const float*)d_in[10], (const float*)d_in[11],
      (const float*)d_in[12], (const float*)d_in[13], (const float*)d_in[14],
      (const float*)d_in[15], (const float*)d_in[16], (const float*)d_in[17],
      (float*)d_out);
}
